// MultiHeadAttention_6932077216113
// MI455X (gfx1250) — matrix operand order repaired and checked by reading, not compiled, not run on hardware
//
#include <hip/hip_runtime.h>
#include <hip/hip_bf16.h>

// MI455X (gfx1250) multi-head attention, wave32 + v_wmma_f32_16x16x32_f16.
// b=4, n=1024, e=768, h=12, d=64. ~32 GFLOP vs ~40MB HBM -> compute bound.
// GEMMs: B tiles staged in LDS (shared by 4 waves) via async-to-LDS double
// buffering (global_load_async_to_lds_b128 + s_wait_asynccnt); A fragments
// and attention operands are register-double-buffered (unconditional wrapped
// prefetch to avoid phi-copy storms) so WMMAs overlap loads.

typedef _Float16 h8  __attribute__((ext_vector_type(8)));
typedef _Float16 h16 __attribute__((ext_vector_type(16)));
typedef float    f8  __attribute__((ext_vector_type(8)));
typedef float    f4  __attribute__((ext_vector_type(4)));
typedef int      i4  __attribute__((ext_vector_type(4)));

#define EMB   768
#define N3E   2304
#define SEQ   1024
#define HEADS 12
#define DH    64
#define BATCH 4
#define ROWS  4096   // BATCH*SEQ
#define BK    32     // K-chunk (WMMA f16 K)

// ---- gfx1250 async global->LDS path (guarded; falls back to plain stores) ----
#if __has_builtin(__builtin_amdgcn_global_load_async_to_lds_b128) && \
    __has_builtin(__builtin_amdgcn_s_wait_asynccnt)
#define HAVE_ASYNC_LDS 1
typedef __attribute__((address_space(1))) i4 as1_i4;
typedef __attribute__((address_space(3))) i4 as3_i4;
__device__ __forceinline__ void cp16_g2l(const _Float16* g, _Float16* l) {
  __builtin_amdgcn_global_load_async_to_lds_b128((as1_i4*)g, (as3_i4*)l, 0, 0);
}
__device__ __forceinline__ void wait_async(int n) {
  if (n == 0) __builtin_amdgcn_s_wait_asynccnt(0);
  else        __builtin_amdgcn_s_wait_asynccnt(2);
}
#else
#define HAVE_ASYNC_LDS 0
__device__ __forceinline__ void cp16_g2l(const _Float16* g, _Float16* l) {
  *(h8*)l = *(const h8*)g;
}
__device__ __forceinline__ void wait_async(int) {}
#endif

__device__ __forceinline__ f8 wmma_f16(h16 a, h16 b, f8 c) {
  return __builtin_amdgcn_wmma_f32_16x16x32_f16(
      /*neg_a=*/false, a, /*neg_b=*/false, b,
      /*c_mod=*/(short)0, c, /*reuse_a=*/false, /*reuse_b=*/false);
}

__device__ __forceinline__ h16 cat16(h8 lo, h8 hi) {
  h16 r;
#pragma unroll
  for (int i = 0; i < 8; ++i) { r[i] = lo[i]; r[8 + i] = hi[i]; }
  return r;
}

// A-matrix 16x32 f16 fragment from a row-major row pointer (&A[row][0]).
// ISA: lanes 0-15 -> VGPR0-3: K=0..7, VGPR4-7: K=16..23; lanes 16-31: +8.
__device__ __forceinline__ h16 load_a_f16(const _Float16* rowp, int k0, int lane) {
  const int kb = (lane & 16) ? 8 : 0;
  h8 lo = *(const h8*)(rowp + k0 + kb);
  h8 hi = *(const h8*)(rowp + k0 + kb + 16);
  return cat16(lo, hi);
}

// B-matrix 32x16 f16 fragment; B stored transposed [N][K] (colp = &BT[n][0]).
// ISA: lanes 0-15 hold K=0..7,16..23, lanes 16-31 hold K=8..15,24..31 (N = lane&15).
__device__ __forceinline__ h16 load_b_f16(const _Float16* colp, int k0, int lane) {
  const int kh = (lane & 16) ? 8 : 0;
  h8 lo = *(const h8*)(colp + k0 + kh);
  h8 hi = *(const h8*)(colp + k0 + kh + 16);
  return cat16(lo, hi);
}

// B fragment from an LDS-staged tile buf[64][BK] (row = n, contiguous K).
__device__ __forceinline__ h16 load_b_lds(const _Float16* buf, int t, int lane) {
  const int kh = (lane & 16) ? 8 : 0;
  const _Float16* p = buf + (t * 16 + (lane & 15)) * BK + kh;
  return cat16(*(const h8*)p, *(const h8*)(p + 16));
}

// A-fragment sourced from an f32 row in LDS (softmax probabilities), cvt->f16.
__device__ __forceinline__ h16 load_a_from_s(const float* rowp, int k0, int lane) {
  const int kb = (lane & 16) ? 8 : 0;
  const f4* p0 = (const f4*)(rowp + k0 + kb);
  const f4* p1 = (const f4*)(rowp + k0 + kb + 16);
  f4 x0 = p0[0], x1 = p0[1], x2 = p1[0], x3 = p1[1];
  h16 r;
#pragma unroll
  for (int i = 0; i < 4; ++i) {
    r[i]      = (_Float16)x0[i];
    r[4 + i]  = (_Float16)x1[i];
    r[8 + i]  = (_Float16)x2[i];
    r[12 + i] = (_Float16)x3[i];
  }
  return r;
}

// Block-cooperative copy of B tile [col0..col0+63][k0..k0+31] -> buf[64][BK].
// 128 threads x 2 b128 = 4KB; with async path this is 2 ASYNC instrs per wave.
__device__ __forceinline__ void copy_btile(const _Float16* wT, int ldk, int col0,
                                           int k0, _Float16* buf, int tid) {
#pragma unroll
  for (int j = 0; j < 2; ++j) {
    const int idx = tid * 2 + j;             // 0..255 over (row, quarter)
    const int row = idx >> 2, q = idx & 3;
    cp16_g2l(wT + (size_t)(col0 + row) * ldk + k0 + q * 8,
             buf + row * BK + q * 8);
  }
}

// Shared GEMM mainloop: acc[4] = A[16 rows x K] * B[K x 64] for one wave,
// B tile double-buffered through LDS, A fragment double-buffered in registers.
__device__ __forceinline__ void gemm_loop(const _Float16* arow, const _Float16* wT,
                                          int K, int col0, int tid, int lane,
                                          _Float16 (*Bt)[64 * BK], f8* acc) {
  const int NK = K / BK;
  copy_btile(wT, K, col0, 0, Bt[0], tid);
  h16 a = load_a_f16(arow, 0, lane);
  for (int kc = 0; kc < NK; ++kc) {
    const bool more  = (kc + 1 < NK);
    const int  knext = more ? (kc + 1) * BK : 0;   // wrap: always in-bounds
    if (more) copy_btile(wT, K, col0, knext, Bt[(kc + 1) & 1], tid);
    wait_async(more ? 2 : 0);      // chunk kc's async copies complete (in-order)
    __syncthreads();               // visible to all 4 waves
    h16 an = load_a_f16(arow, knext, lane);        // unconditional prefetch
    const _Float16* buf = Bt[kc & 1];
    h16 b0 = load_b_lds(buf, 0, lane);             // distinct regs -> partial
    h16 b1 = load_b_lds(buf, 1, lane);             // s_wait_dscnt overlap
    h16 b2 = load_b_lds(buf, 2, lane);
    h16 b3 = load_b_lds(buf, 3, lane);
    acc[0] = wmma_f16(a, b0, acc[0]);
    acc[1] = wmma_f16(a, b1, acc[1]);
    acc[2] = wmma_f16(a, b2, acc[2]);
    acc[3] = wmma_f16(a, b3, acc[3]);
    a = an;
    __syncthreads();               // reads done before buf re-written
  }
}

// ---------------- conversion / layout kernels ----------------

__global__ void k_cvt_f16(const float* __restrict__ in, _Float16* __restrict__ out, int n) {
  int i = blockIdx.x * blockDim.x + threadIdx.x;
  int stride = gridDim.x * blockDim.x;
  for (; i < n; i += stride) out[i] = (_Float16)in[i];
}

// out[c*rows + r] = (f16) in[r*cols + c]  (weights -> [N][K] for B fragments)
__global__ void k_transpose_f16(const float* __restrict__ in, _Float16* __restrict__ out,
                                int rows, int cols) {
  int i = blockIdx.x * blockDim.x + threadIdx.x;
  int stride = gridDim.x * blockDim.x;
  int n = rows * cols;
  for (; i < n; i += stride) {
    int r = i / cols, c = i - r * cols;
    out[(size_t)c * rows + r] = (_Float16)in[i];
  }
}

// ---------------- QKV projection GEMM ----------------
// C[4096,2304] = Xh @ Wqkv + bqkv, scattered to Q/K/V per head.
__global__ void k_gemm_qkv(const _Float16* __restrict__ xh,
                           const _Float16* __restrict__ wT,   // [2304][768]
                           const float* __restrict__ bias,
                           _Float16* __restrict__ qb,         // [bh][n][d]
                           _Float16* __restrict__ kb,         // [bh][n][d]
                           _Float16* __restrict__ vT) {       // [bh][d][n]
  __shared__ _Float16 Bt[2][64 * BK];                          // 8 KB
  const int tid  = threadIdx.x;
  const int lane = tid & 31;
  const int wv   = tid >> 5;
  const int row0 = blockIdx.y * 64 + wv * 16;
  const int col0 = blockIdx.x * 64;
  const int nn   = lane & 15;
  const int mb   = (lane & 16) ? 8 : 0;

  f8 acc[4] = {};
  gemm_loop(xh + (size_t)(row0 + nn) * EMB, wT, EMB, col0, tid, lane, Bt, acc);

#pragma unroll
  for (int t = 0; t < 4; ++t) {
    const int c   = col0 + t * 16 + nn;        // column in [0,2304)
    const int sel = c % 3;                      // 0=Q 1=K 2=V
    const int hh  = c / 192;                    // head
    const int dd  = (c % 192) / 3;              // d within head
    const float bv = bias[c];
#pragma unroll
    for (int i = 0; i < 8; ++i) {
      const int m  = row0 + mb + i;
      const int b_ = m >> 10, n_ = m & 1023;
      const size_t bh = (size_t)(b_ * HEADS + hh);
      const _Float16 hv = (_Float16)(acc[t][i] + bv);
      if (sel == 0)       qb[(bh * SEQ + n_) * DH + dd] = hv;
      else if (sel == 1)  kb[(bh * SEQ + n_) * DH + dd] = hv;
      else                vT[(bh * DH + dd) * SEQ + n_] = hv;
    }
  }
}

// ---------------- attention (per bh, 16-query-row tile, 1 wave) ----------------
// LDS: S[16][1024] f32 = 64KB (dynamic). S = QK^T -> softmax/sqrt(768) -> O = P V.
__global__ void k_attn(const _Float16* __restrict__ qb,
                       const _Float16* __restrict__ kb,
                       const _Float16* __restrict__ vT,
                       _Float16* __restrict__ oh) {            // [b][n][768]
  extern __shared__ float S[];                                  // [16][1024]
  const int lane  = threadIdx.x & 31;
  const int bh    = blockIdx.y;                                 // 0..47
  const int qrow0 = blockIdx.x * 16;
  const int nn    = lane & 15;
  const int mb    = (lane & 16) ? 8 : 0;

  const _Float16* qhead = qb + (size_t)bh * SEQ * DH;
  const _Float16* khead = kb + (size_t)bh * SEQ * DH;
  const _Float16* vhead = vT + (size_t)bh * DH * SEQ;

  // Phase 1: S = Q K^T ; K-tile fragments register-double-buffered (wrapped).
  const _Float16* qrowp = qhead + (size_t)(qrow0 + nn) * DH;
  h16 a0 = load_a_f16(qrowp, 0, lane);
  h16 a1 = load_a_f16(qrowp, 32, lane);
  {
    const _Float16* kr = khead + (size_t)nn * DH;
    h16 b0 = load_b_f16(kr, 0, lane);
    h16 b1 = load_b_f16(kr, 32, lane);
    for (int nt = 0; nt < SEQ / 16; ++nt) {
      const int ntn = (nt + 1) & (SEQ / 16 - 1);               // wrap prefetch
      const _Float16* krn = khead + (size_t)(ntn * 16 + nn) * DH;
      h16 nb0 = load_b_f16(krn, 0, lane);
      h16 nb1 = load_b_f16(krn, 32, lane);
      f8 c = {};
      c = wmma_f16(a0, b0, c);
      c = wmma_f16(a1, b1, c);
#pragma unroll
      for (int i = 0; i < 8; ++i)
        S[(mb + i) * SEQ + nt * 16 + nn] = c[i];
      b0 = nb0; b1 = nb1;
    }
  }
  __syncthreads();

  // Phase 2: row softmax; lane covers row lane&15, half (lane>>4) of 1024 cols.
  // Folds the reference's post-softmax 1/sqrt(768) scale.
  {
    float* row = S + nn * SEQ + ((lane & 16) ? 512 : 0);
    float m = -3.0e38f;
    for (int j = 0; j < 512; ++j) m = fmaxf(m, row[j]);
    m = fmaxf(m, __shfl_xor(m, 16));
    float s = 0.0f;
    for (int j = 0; j < 512; ++j) { float e = __expf(row[j] - m); row[j] = e; s += e; }
    s += __shfl_xor(s, 16);
    const float scale = 1.0f / (s * 27.712812921102035f);       // sqrt(768)
    for (int j = 0; j < 512; ++j) row[j] *= scale;
  }
  __syncthreads();

  // Phase 3: O[16,64] = P[16,1024] @ V[1024,64] (V transposed [d][n]).
  // Next chunk's P and V fragments prefetched into registers (wrapped).
  f8 o[4] = {};
  const float* prow = S + nn * SEQ;
  h16 a = load_a_from_s(prow, 0, lane);
  h16 bv0 = load_b_f16(vhead + (size_t)(0 * 16 + nn) * SEQ, 0, lane);
  h16 bv1 = load_b_f16(vhead + (size_t)(1 * 16 + nn) * SEQ, 0, lane);
  h16 bv2 = load_b_f16(vhead + (size_t)(2 * 16 + nn) * SEQ, 0, lane);
  h16 bv3 = load_b_f16(vhead + (size_t)(3 * 16 + nn) * SEQ, 0, lane);
  for (int k0 = 0; k0 < SEQ; k0 += BK) {
    const int kn = (k0 + BK) & (SEQ - 1);                       // wrap prefetch
    h16 an  = load_a_from_s(prow, kn, lane);
    h16 nb0 = load_b_f16(vhead + (size_t)(0 * 16 + nn) * SEQ, kn, lane);
    h16 nb1 = load_b_f16(vhead + (size_t)(1 * 16 + nn) * SEQ, kn, lane);
    h16 nb2 = load_b_f16(vhead + (size_t)(2 * 16 + nn) * SEQ, kn, lane);
    h16 nb3 = load_b_f16(vhead + (size_t)(3 * 16 + nn) * SEQ, kn, lane);
    o[0] = wmma_f16(a, bv0, o[0]);
    o[1] = wmma_f16(a, bv1, o[1]);
    o[2] = wmma_f16(a, bv2, o[2]);
    o[3] = wmma_f16(a, bv3, o[3]);
    a = an; bv0 = nb0; bv1 = nb1; bv2 = nb2; bv3 = nb3;
  }

  const int b_ = bh / HEADS, hh = bh % HEADS;
#pragma unroll
  for (int t = 0; t < 4; ++t) {
    const int dd = t * 16 + nn;
#pragma unroll
    for (int i = 0; i < 8; ++i) {
      const int n_ = qrow0 + mb + i;
      oh[((size_t)b_ * SEQ + n_) * EMB + hh * DH + dd] = (_Float16)o[t][i];
    }
  }
}

// ---------------- output projection GEMM ----------------
__global__ void k_gemm_proj(const _Float16* __restrict__ oh,
                            const _Float16* __restrict__ wT,  // [768][768]
                            const float* __restrict__ bias,
                            float* __restrict__ out) {
  __shared__ _Float16 Bt[2][64 * BK];
  const int tid  = threadIdx.x;
  const int lane = tid & 31;
  const int wv   = tid >> 5;
  const int row0 = blockIdx.y * 64 + wv * 16;
  const int col0 = blockIdx.x * 64;
  const int nn   = lane & 15;
  const int mb   = (lane & 16) ? 8 : 0;

  f8 acc[4] = {};
  gemm_loop(oh + (size_t)(row0 + nn) * EMB, wT, EMB, col0, tid, lane, Bt, acc);

#pragma unroll
  for (int t = 0; t < 4; ++t) {
    const int c = col0 + t * 16 + nn;
    const float bv = bias[c];
#pragma unroll
    for (int i = 0; i < 8; ++i) {
      const int m = row0 + mb + i;
      out[(size_t)m * EMB + c] = acc[t][i] + bv;
    }
  }
}

// ---------------- host launcher ----------------

extern "C" void kernel_launch(void* const* d_in, const int* in_sizes, int n_in,
                              void* d_out, int out_size, void* d_ws, size_t ws_size,
                              hipStream_t stream) {
  (void)in_sizes; (void)n_in; (void)out_size; (void)ws_size;
  const float* x     = (const float*)d_in[0];   // [4,1024,768]
  const float* Wqkv  = (const float*)d_in[1];   // [768,2304]
  const float* bqkv  = (const float*)d_in[2];   // [2304]
  const float* Wproj = (const float*)d_in[3];   // [768,768]
  const float* bproj = (const float*)d_in[4];   // [768]
  float* out = (float*)d_out;                   // [4,1024,768] f32

  _Float16* ws = (_Float16*)d_ws;
  const size_t XH_N = (size_t)ROWS * EMB;
  const size_t WQ_N = (size_t)N3E * EMB;
  const size_t WP_N = (size_t)EMB * EMB;
  _Float16* xh     = ws;
  _Float16* wqkvT  = xh + XH_N;
  _Float16* wprojT = wqkvT + WQ_N;
  _Float16* qb     = wprojT + WP_N;
  _Float16* kb     = qb + XH_N;
  _Float16* vT     = kb + XH_N;
  _Float16* oh     = vT + XH_N;   // total ~36.2 MB

  k_cvt_f16<<<2048, 256, 0, stream>>>(x, xh, (int)XH_N);
  k_transpose_f16<<<2048, 256, 0, stream>>>(Wqkv, wqkvT, EMB, N3E);
  k_transpose_f16<<<1024, 256, 0, stream>>>(Wproj, wprojT, EMB, EMB);

  k_gemm_qkv<<<dim3(N3E / 64, ROWS / 64), 128, 0, stream>>>(xh, wqkvT, bqkv, qb, kb, vT);

  k_attn<<<dim3(SEQ / 16, BATCH * HEADS), 32, 16 * SEQ * sizeof(float), stream>>>(qb, kb, vT, oh);

  k_gemm_proj<<<dim3(EMB / 64, ROWS / 64), 128, 0, stream>>>(oh, wprojT, bproj, out);
}
